// GPT2Synapse_38044820308021
// MI455X (gfx1250) — matrix operand order repaired and checked by reading, not compiled, not run on hardware
//
#include <hip/hip_runtime.h>
#include <math.h>

// ---------------- problem constants ----------------
#define Bb 8
#define Tt 1024
#define Dd 768
#define Hh 12
#define HDh 64
#define WIN 768
#define MM (Bb * Tt)          // 8192 rows

typedef __attribute__((ext_vector_type(16))) __bf16 v16bf;
typedef __attribute__((ext_vector_type(8)))  float  v8f;
typedef __attribute__((ext_vector_type(4)))  unsigned int u32x4;

// ---------------- helpers ----------------
__device__ __forceinline__ __bf16 f2bf(float f) { return (__bf16)f; }

union V16U { v16bf v; struct { u32x4 lo, hi; } u; };

// 16 contiguous bf16 (32B, 16B-aligned) -> v16bf
__device__ __forceinline__ v16bf ld16(const __bf16* p) {
    V16U t;
    t.u.lo = *(const u32x4*)p;
    t.u.hi = *(const u32x4*)(p + 8);
    return t.v;
}
// A-fragment per-lane pattern: 16 contiguous bf16 at p[0..15]
__device__ __forceinline__ v16bf ldA(const __bf16* p) {
    V16U t;
    t.u.lo = *(const u32x4*)p;
    t.u.hi = *(const u32x4*)(p + 8);
    return t.v;
}

__device__ __forceinline__ v8f wmma_bf16(v16bf a, v16bf b, v8f c) {
    return __builtin_amdgcn_wmma_f32_16x16x32_bf16(
        /*neg_a=*/false, a, /*neg_b=*/false, b,
        /*c_mod=*/(short)0, c, /*reuse_a=*/false, /*reuse_b=*/false);
}

// async global->LDS copy of 32 bytes per lane (two b128 issues, ASYNCcnt-tracked)
__device__ __forceinline__ void async_copy32(unsigned lds_off, const __bf16* gaddr) {
    asm volatile(
        "global_load_async_to_lds_b128 %0, %1, off\n\t"
        "global_load_async_to_lds_b128 %0, %1, off offset:16"
        :: "v"(lds_off), "v"(gaddr) : "memory");
}
__device__ __forceinline__ void async_wait0() {
    asm volatile("s_wait_asynccnt 0" ::: "memory");
}

// ---------------- fp32 [K,N] -> bf16 [N,K] tiled transpose-convert ----------------
__global__ __launch_bounds__(256) void cvt_t_kernel(const float* __restrict__ in,
                                                    __bf16* __restrict__ out,
                                                    int K, int N) {
    __shared__ __bf16 tile[32 * 40];
    const int bk = blockIdx.y * 32, bn = blockIdx.x * 32;
    const int t = threadIdx.x;
    const int r = t >> 3, c4 = (t & 7) * 4;
    // coalesced read of in[bk+r][bn+c4..+3]
    const float* src = in + (size_t)(bk + r) * N + bn + c4;
#pragma unroll
    for (int i = 0; i < 4; i++) tile[r * 40 + c4 + i] = f2bf(src[i]);
    __syncthreads();
    // coalesced write of out[bn+r][bk+c4..+3]
    __bf16* dst = out + (size_t)(bn + r) * K + bk + c4;
#pragma unroll
    for (int i = 0; i < 4; i++) dst[i] = tile[(c4 + i) * 40 + r];
}

// ---------------- V transpose: qkv[b,t,2D+h*64+hd] -> Vt[(b*H+h), hd, t] ----------------
__global__ __launch_bounds__(256) void vtrans_kernel(const __bf16* __restrict__ qkv,
                                                     __bf16* __restrict__ Vt) {
    __shared__ __bf16 tile[64 * 72];
    int idx = blockIdx.x;
    const int kt = idx & 15; idx >>= 4;       // T/64 = 16 key tiles
    const int h = idx % Hh;
    const int b = idx / Hh;
    const int key0 = kt * 64;
    const int t = threadIdx.x;
    const int r = t >> 2, c = (t & 3) * 16;

    // read 64 keys x 64 hd (coalesced, 32B per thread)
    const __bf16* src = qkv + (size_t)(b * Tt + key0 + r) * (3 * Dd) + 2 * Dd + h * HDh + c;
    __bf16* sp = &tile[r * 72 + c];
    *(u32x4*)sp = *(const u32x4*)src;
    *(u32x4*)(sp + 8) = *(const u32x4*)(src + 8);
    __syncthreads();

    // write transposed: Vt[(b*H+h), hd=r, key0+c..+15] (coalesced 32B)
    __bf16 vals[16];
#pragma unroll
    for (int i = 0; i < 16; i++) vals[i] = tile[(c + i) * 72 + r];
    __bf16* dst = Vt + ((size_t)(b * Hh + h) * HDh + r) * Tt + key0 + c;
    *(u32x4*)dst = *(const u32x4*)&vals[0];
    *(u32x4*)(dst + 8) = *(const u32x4*)&vals[8];
}

// ---------------- LayerNorm (row of D=768) -> bf16 ----------------
__global__ __launch_bounds__(256) void ln_kernel(const float* __restrict__ x,
                                                 const float* __restrict__ w,
                                                 const float* __restrict__ b,
                                                 __bf16* __restrict__ out) {
    __shared__ float red[256];
    const int tid = threadIdx.x;
    const size_t row = blockIdx.x;
    const float* xr = x + row * Dd;

    float s = 0.f;
    for (int t = tid; t < Dd; t += 256) s += xr[t];
    red[tid] = s; __syncthreads();
    for (int o = 128; o > 0; o >>= 1) { if (tid < o) red[tid] += red[tid + o]; __syncthreads(); }
    const float mu = red[0] * (1.0f / Dd);
    __syncthreads();

    float v = 0.f;
    for (int t = tid; t < Dd; t += 256) { float d = xr[t] - mu; v += d * d; }
    red[tid] = v; __syncthreads();
    for (int o = 128; o > 0; o >>= 1) { if (tid < o) red[tid] += red[tid + o]; __syncthreads(); }
    const float rstd = rsqrtf(red[0] * (1.0f / Dd) + 1e-5f);

    __bf16* op = out + row * Dd;
    for (int t = tid; t < Dd; t += 256)
        op[t] = f2bf((xr[t] - mu) * rstd * w[t] + b[t]);
}

// ---------------- WMMA GEMM: C[M,N] = epi(A[M,K] @ Wt[N,K]^T + bias) ----------------
// Both operands K-major; LDS pitch 40 bf16 (80B, 16B-aligned chunks).
// Staging uses gfx1250 async global->LDS DMA (ASYNCcnt).
#define BM 128
#define BN 128
#define BK 32
#define LDP 40

#define EPI_BF16_BIAS 0
#define EPI_RESID_F32 1
#define EPI_GELU_BF16 2

template<int EPI>
__global__ __launch_bounds__(256) void gemm_bf16_kernel(
    const __bf16* __restrict__ A, const __bf16* __restrict__ Wt,
    const float* __restrict__ bias, const float* __restrict__ resid,
    __bf16* __restrict__ outB, float* __restrict__ outF,
    int M, int N, int K)
{
    __shared__ __bf16 As[BM * LDP];
    __shared__ __bf16 Bs[BN * LDP];

    const int tid  = threadIdx.x;
    const int lane = tid & 31;
    const int wid  = tid >> 5;
    const int bm = blockIdx.y * BM;
    const int bn = blockIdx.x * BN;
    const int wm = (wid >> 1) * 32;     // 4 waves along M
    const int wn = (wid & 1) * 64;      // 2 waves along N

    const int mrow = lane & 15;
    const int ncol = lane & 15;
    const int kgrp = (lane >> 4) * 16;  // A-frag K base
    const int kb16 = (lane >> 4) * 16;  // B-frag K base

    // staging assignment: one 16-elt (32B) run per thread per tile
    const int sr = tid >> 1, sk = (tid & 1) * 16;
    const __bf16* ga = A  + (size_t)(bm + sr) * K + sk;
    const __bf16* gb = Wt + (size_t)(bn + sr) * K + sk;
    const unsigned ldsA = (unsigned)(size_t)&As[sr * LDP + sk];
    const unsigned ldsB = (unsigned)(size_t)&Bs[sr * LDP + sk];

    v8f acc[2][4] = {};

    for (int k0 = 0; k0 < K; k0 += BK) {
        async_copy32(ldsA, ga + k0);
        async_copy32(ldsB, gb + k0);
        async_wait0();
        __syncthreads();

        v16bf af[2], bf[4];
#pragma unroll
        for (int mi = 0; mi < 2; mi++)
            af[mi] = ldA(&As[(wm + mi * 16 + mrow) * LDP + kgrp]);
#pragma unroll
        for (int ni = 0; ni < 4; ni++)
            bf[ni] = ld16(&Bs[(wn + ni * 16 + ncol) * LDP + kb16]);
#pragma unroll
        for (int mi = 0; mi < 2; mi++)
#pragma unroll
            for (int ni = 0; ni < 4; ni++)
                acc[mi][ni] = wmma_bf16(af[mi], bf[ni], acc[mi][ni]);
        __syncthreads();
    }

    // epilogue (C layout: M = r + 8*(lane>=16), N = lane&15)
#pragma unroll
    for (int mi = 0; mi < 2; mi++)
#pragma unroll
    for (int ni = 0; ni < 4; ni++) {
#pragma unroll
        for (int r = 0; r < 8; r++) {
            const int row = bm + wm + mi * 16 + r + ((lane >> 4) << 3);
            const int col = bn + wn + ni * 16 + (lane & 15);
            float v = acc[mi][ni][r] + bias[col];
            const size_t idx = (size_t)row * N + col;
            if (EPI == EPI_BF16_BIAS) {
                outB[idx] = f2bf(v);
            } else if (EPI == EPI_GELU_BF16) {
                outB[idx] = f2bf(0.5f * v * (1.0f + erff(v * 0.70710678118654752f)));
            } else {
                outF[idx] = resid[idx] + v;
            }
        }
    }
}

// ---------------- flash attention: one wave per (b, h, 16-query tile) ----------------
__global__ __launch_bounds__(32) void attn_kernel(const __bf16* __restrict__ qkv,
                                                  const __bf16* __restrict__ Vt,
                                                  const int* __restrict__ amask,
                                                  __bf16* __restrict__ out)
{
    __shared__ __bf16 psh[16 * 40];   // 16x32 P tile, 80B pitch (16B aligned)

    const int lane = threadIdx.x;
    int idx = blockIdx.x;
    const int qt = idx & 63; idx >>= 6;            // T/16 = 64 query tiles
    const int h  = idx % Hh;
    const int b  = idx / Hh;
    const int qbase = qt * 16;

    const size_t rs = 3 * Dd;                      // qkv row stride
    const __bf16* qp = qkv + (size_t)b * Tt * rs + h * HDh;
    const __bf16* kp = qp + Dd;
    const __bf16* vtp = Vt + (size_t)(b * Hh + h) * HDh * Tt;

    const int m    = lane & 15;
    const int kg   = (lane >> 4) * 16;
    const int ncol = lane & 15;
    const int kb16 = (lane >> 4) * 16;

    // Q fragments for the two HD k-steps (16x32 each)
    v16bf aq[2];
#pragma unroll
    for (int ks = 0; ks < 2; ks++)
        aq[ks] = ldA(qp + (size_t)(qbase + m) * rs + ks * 32 + kg);

    float mrun[8], lrun[8];
    v8f o[4] = {};
#pragma unroll
    for (int r = 0; r < 8; r++) { mrun[r] = -3.0e38f; lrun[r] = 0.f; }

    int lo = qbase - (WIN - 1); if (lo < 0) lo = 0;
    const int kc_lo = lo & ~31;
    const float scale = 0.125f;   // 1/sqrt(64)

    for (int kc = kc_lo; kc <= qbase + 15; kc += 32) {
        // ---- scores: two 16x16 key tiles, accumulate over HD=64 ----
        v8f s[2];
#pragma unroll
        for (int nt = 0; nt < 2; nt++) {
            const __bf16* kr = kp + (size_t)(kc + nt * 16 + ncol) * rs;
            v8f sc = {};
            sc = wmma_bf16(aq[0], ld16(kr + kb16), sc);
            sc = wmma_bf16(aq[1], ld16(kr + 32 + kb16), sc);
            s[nt] = sc;
        }

        const int key0 = kc + ncol, key1 = kc + 16 + ncol;
        const int am0 = amask[b * Tt + key0];
        const int am1 = amask[b * Tt + key1];

        // ---- online softmax over the 32-key chunk ----
#pragma unroll
        for (int r = 0; r < 8; r++) {
            const int qrow = qbase + r + ((lane >> 4) << 3);
            bool k0ok = (key0 <= qrow) && (key0 >= qrow - (WIN - 1)) && (am0 != 0);
            bool k1ok = (key1 <= qrow) && (key1 >= qrow - (WIN - 1)) && (am1 != 0);
            float s0 = k0ok ? s[0][r] * scale : -3.0e38f;
            float s1 = k1ok ? s[1][r] * scale : -3.0e38f;

            float v = fmaxf(s0, s1);
#pragma unroll
            for (int off = 1; off < 16; off <<= 1) v = fmaxf(v, __shfl_xor(v, off, 32));
            const float nm = fmaxf(mrun[r], v);
            const float cf = expf(mrun[r] - nm);
            mrun[r] = nm;

            const float p0 = expf(s0 - nm);
            const float p1 = expf(s1 - nm);
            s[0][r] = p0; s[1][r] = p1;

            float ps = p0 + p1;
#pragma unroll
            for (int off = 1; off < 16; off <<= 1) ps += __shfl_xor(ps, off, 32);
            lrun[r] = lrun[r] * cf + ps;
#pragma unroll
            for (int nt = 0; nt < 4; nt++) o[nt][r] *= cf;
        }

        // ---- transpose P (C-layout -> A-fragment layout) via LDS ----
        __syncthreads();
#pragma unroll
        for (int r = 0; r < 8; r++) {
            const int mm = r + ((lane >> 4) << 3);
            psh[mm * 40 + ncol]      = f2bf(s[0][r]);
            psh[mm * 40 + 16 + ncol] = f2bf(s[1][r]);
        }
        __syncthreads();
        const v16bf pa = ldA(&psh[m * 40 + kg]);

        // ---- P @ V over 4 hd tiles (Vt is [hd][key], contiguous along key) ----
#pragma unroll
        for (int nt = 0; nt < 4; nt++) {
            const __bf16* vr = vtp + (size_t)(nt * 16 + ncol) * Tt + kc + kb16;
            o[nt] = wmma_bf16(pa, ld16(vr), o[nt]);
        }
    }

    // ---- finalize and store [B,T,D] bf16 ----
#pragma unroll
    for (int r = 0; r < 8; r++) {
        const float inv = 1.0f / lrun[r];
        const int row = qbase + r + ((lane >> 4) << 3);
        __bf16* op = out + ((size_t)(b * Tt + row)) * Dd + h * HDh;
#pragma unroll
        for (int nt = 0; nt < 4; nt++)
            op[nt * 16 + ncol] = f2bf(o[nt][r] * inv);
    }
}

// ---------------- launch ----------------
extern "C" void kernel_launch(void* const* d_in, const int* in_sizes, int n_in,
                              void* d_out, int out_size, void* d_ws, size_t ws_size,
                              hipStream_t stream) {
    (void)in_sizes; (void)n_in; (void)out_size; (void)ws_size;

    const float* x       = (const float*)d_in[0];
    const int*   amask   = (const int*)d_in[1];
    const float* ln1_w   = (const float*)d_in[2];
    const float* ln1_b   = (const float*)d_in[3];
    const float* w_attn  = (const float*)d_in[4];
    const float* b_attn  = (const float*)d_in[5];
    const float* w_proj  = (const float*)d_in[6];
    const float* b_proj  = (const float*)d_in[7];
    const float* ln2_w   = (const float*)d_in[8];
    const float* ln2_b   = (const float*)d_in[9];
    const float* w_fc    = (const float*)d_in[10];
    const float* b_fc    = (const float*)d_in[11];
    const float* w_fcp   = (const float*)d_in[12];
    const float* b_fcp   = (const float*)d_in[13];
    float* outp = (float*)d_out;

    char* ws = (char*)d_ws;
    size_t off = 0;
    auto alloc = [&](size_t bytes) { size_t o = off; off = (off + bytes + 255) & ~(size_t)255; return o; };

    // transposed bf16 weights [N,K]
    __bf16* wbA  = (__bf16*)(ws + alloc((size_t)Dd * 3 * Dd * 2));
    __bf16* wbP  = (__bf16*)(ws + alloc((size_t)Dd * Dd * 2));
    __bf16* wbF  = (__bf16*)(ws + alloc((size_t)Dd * 4 * Dd * 2));
    __bf16* wbFP = (__bf16*)(ws + alloc((size_t)4 * Dd * Dd * 2));
    __bf16* x1b  = (__bf16*)(ws + alloc((size_t)MM * Dd * 2));
    __bf16* qkvb = (__bf16*)(ws + alloc((size_t)MM * 3 * Dd * 2));
    __bf16* Vt   = (__bf16*)(ws + alloc((size_t)Bb * Hh * HDh * Tt * 2));
    __bf16* attb = (__bf16*)(ws + alloc((size_t)MM * Dd * 2));
    float*  xmid = (float*) (ws + alloc((size_t)MM * Dd * 4));
    __bf16* x3b  = (__bf16*)(ws + alloc((size_t)MM * Dd * 2));
    __bf16* hb   = (__bf16*)(ws + alloc((size_t)MM * 4 * Dd * 2));

    // weight transpose-convert fp32 [K,N] -> bf16 [N,K]
    cvt_t_kernel<<<dim3(3 * Dd / 32, Dd / 32),     256, 0, stream>>>(w_attn, wbA,  Dd, 3 * Dd);
    cvt_t_kernel<<<dim3(Dd / 32, Dd / 32),         256, 0, stream>>>(w_proj, wbP,  Dd, Dd);
    cvt_t_kernel<<<dim3(4 * Dd / 32, Dd / 32),     256, 0, stream>>>(w_fc,   wbF,  Dd, 4 * Dd);
    cvt_t_kernel<<<dim3(Dd / 32, 4 * Dd / 32),     256, 0, stream>>>(w_fcp,  wbFP, 4 * Dd, Dd);

    // LN1: x -> x1 (bf16)
    ln_kernel<<<MM, 256, 0, stream>>>(x, ln1_w, ln1_b, x1b);

    // QKV GEMM: [8192,768] x [768,2304] + bias -> bf16
    gemm_bf16_kernel<EPI_BF16_BIAS><<<dim3(3 * Dd / BN, MM / BM), 256, 0, stream>>>(
        x1b, wbA, b_attn, nullptr, qkvb, nullptr, MM, 3 * Dd, Dd);

    // V transpose for the P@V stage
    vtrans_kernel<<<Bb * Hh * (Tt / 64), 256, 0, stream>>>(qkvb, Vt);

    // windowed-causal flash attention
    attn_kernel<<<Bb * Hh * (Tt / 16), 32, 0, stream>>>(qkvb, Vt, amask, attb);

    // proj GEMM + residual: xmid = x + attn @ w_proj + b_proj  (fp32)
    gemm_bf16_kernel<EPI_RESID_F32><<<dim3(Dd / BN, MM / BM), 256, 0, stream>>>(
        attb, wbP, b_proj, x, nullptr, xmid, MM, Dd, Dd);

    // LN2: xmid -> x3 (bf16)
    ln_kernel<<<MM, 256, 0, stream>>>(xmid, ln2_w, ln2_b, x3b);

    // FC GEMM + exact GELU: h = gelu(x3 @ w_fc + b_fc) (bf16)
    gemm_bf16_kernel<EPI_GELU_BF16><<<dim3(4 * Dd / BN, MM / BM), 256, 0, stream>>>(
        x3b, wbF, b_fc, nullptr, hb, nullptr, MM, 4 * Dd, Dd);

    // FC-proj GEMM + residual -> d_out (fp32): out = xmid + h @ w_fc_proj + b
    gemm_bf16_kernel<EPI_RESID_F32><<<dim3(Dd / BN, MM / BM), 256, 0, stream>>>(
        hb, wbFP, b_fcp, xmid, nullptr, outp, MM, Dd, 4 * Dd);
}
